// Net_62620623175884
// MI455X (gfx1250) — matrix operand order repaired and checked by reading, not compiled, not run on hardware
//
#include <hip/hip_runtime.h>
#include <hip/hip_bf16.h>

// ---------------------------------------------------------------------------
// BrainGNN-style EGAT network, dense reformulation for fully-connected graphs.
// B=64 graphs, N=128 nodes, K=16 pooled nodes, H=30, HEADS=5 (first layer).
// Heavy op: per-graph attention aggregation  out = alpha^T @ h  done with
// v_wmma_f32_16x16x32_f16 (f16 in, f32 accumulate). h is staged transposed in
// LDS so both A and B fragments are contiguous 128-bit DS loads. The
// multi-head (atomic accumulate) vs single-head (plain store) path is a
// template parameter so store loops compile branch-free.
// ---------------------------------------------------------------------------

typedef __attribute__((ext_vector_type(16))) _Float16 v16h;
typedef __attribute__((ext_vector_type(8)))  float    v8f;

#define NEG_SLOPE 0.2f

static inline unsigned cdiv(size_t a, size_t b) { return (unsigned)((a + b - 1) / b); }

// ---------------- WMMA fragment loaders (contiguous k order, same for A and B) -----------
// A: 16x32 f16, row-major in LDS. lane<16: K 0..15; lane>=16: K 16..31.
static __device__ __forceinline__ v16h lds_load_a16(const _Float16* A, int lda,
                                                    int m0, int k0, int lane) {
  int m = m0 + (lane & 15);
  int koff = (lane >> 4) << 4;
  const _Float16* row = A + (size_t)m * lda + k0;
  v16h a;
#pragma unroll
  for (int j = 0; j < 8; ++j) a[j] = row[koff + j];
#pragma unroll
  for (int j = 0; j < 8; ++j) a[8 + j] = row[8 + koff + j];
  return a;
}
// B fragment from B^T stored row-major [N][K] in LDS (contiguous per lane).
// B layout: lane<16: col n=lane, K 0..15; lane>=16: col n=lane-16, K 16..31.
static __device__ __forceinline__ v16h lds_load_bT16(const _Float16* BT, int ldb,
                                                     int k0, int n0, int lane) {
  int n = n0 + (lane & 15);
  int koff = (lane >> 4) << 4;
  const _Float16* row = BT + (size_t)n * ldb + k0 + koff;
  v16h b;
#pragma unroll
  for (int j = 0; j < 16; ++j) b[j] = row[j];
  return b;
}

// ---------------- generic helpers ------------------------------------------
__global__ void k_fill(float* p, float v, size_t n) {
  size_t i = (size_t)blockIdx.x * blockDim.x + threadIdx.x;
  if (i < n) p[i] = v;
}

// y[r, 0..cout) = x[r, 0..cin) @ W + bias
__global__ void k_linear(const float* __restrict__ x, int ldx,
                         const float* __restrict__ W, const float* __restrict__ bias,
                         float* __restrict__ y, int ldy, int rows, int cin, int cout) {
  int idx = blockIdx.x * blockDim.x + threadIdx.x;
  if (idx >= rows * cout) return;
  int r = idx / cout, j = idx - r * cout;
  float acc = bias ? bias[j] : 0.0f;
  const float* xr = x + (size_t)r * ldx;
  for (int i = 0; i < cin; ++i) acc += xr[i] * W[i * cout + j];
  y[(size_t)r * ldy + j] = acc;
}

// as/ad[n,hd] = sum_c h[n,hd,c] * att_{src,dst}[hd,c]
__global__ void k_attn_dots(const float* __restrict__ h,
                            const float* __restrict__ asrc, const float* __restrict__ adst,
                            float* __restrict__ as_o, float* __restrict__ ad_o,
                            int NT, int heads, int cout) {
  int idx = blockIdx.x * blockDim.x + threadIdx.x;
  if (idx >= NT * heads) return;
  int n = idx / heads, hd = idx - n * heads;
  const float* hp = h + ((size_t)n * heads + hd) * cout;
  float s1 = 0.0f, s2 = 0.0f;
  for (int c = 0; c < cout; ++c) {
    s1 += hp[c] * asrc[hd * cout + c];
    s2 += hp[c] * adst[hd * cout + c];
  }
  as_o[idx] = s1;
  ad_o[idx] = s2;
}

// ---------------- fused EGAT attention + WMMA aggregation -------------------
// One block per (graph, head). Dense column-softmax over incoming edges, then
// out = alpha^T @ h via v_wmma_f32_16x16x32_f16. MH=true accumulates the head
// mean via atomics; MH=false uses plain stores (unique writers).
template <int NODES, bool MH>
__global__ void k_egat(const float* __restrict__ h,
                       const float* __restrict__ as_in, const float* __restrict__ ad_in,
                       const float* __restrict__ ea_in,
                       float* __restrict__ alpha_out, float* __restrict__ out_acc,
                       int heads, int cout) {
  constexpr int KP = (NODES < 32) ? 32 : NODES;   // K padded to multiple of 32
  constexpr int KPS = KP + 8;                     // LDS row strides
  __shared__ _Float16 shT[32 * KPS];              // h^T: [32 channels][KP nodes]
  __shared__ _Float16 sAT[NODES * KPS];           // alpha^T [d][s]
  __shared__ float s_as[NODES], s_ad[NODES], s_mx[NODES], s_dn[NODES];

  const int b = blockIdx.x, hd = blockIdx.y;
  const int tid = threadIdx.x, lane = tid & 31, wv = tid >> 5, nwv = blockDim.x >> 5;
  const int E = NODES * NODES;
  const float inv_h = 1.0f / (float)heads;
  const int hstride = heads * cout;
  const float* hb = h + (size_t)b * NODES * hstride + hd * cout;
  const float* ea = ea_in + (size_t)b * E;

  // stage h^T (f32 -> f16) into LDS: shT[c][s], zero pad to 32 ch / KP nodes
  for (int i = tid; i < 32 * KP; i += blockDim.x) {
    int c = i / KP, s = i - c * KP;
    float v = (s < NODES && c < cout) ? hb[(size_t)s * hstride + c] : 0.0f;
    shT[c * KPS + s] = (_Float16)v;
  }
  for (int i = tid; i < NODES; i += blockDim.x) {
    s_as[i] = as_in[(b * NODES + i) * heads + hd];
    s_ad[i] = ad_in[(b * NODES + i) * heads + hd];
  }
  __syncthreads();

  // column softmax stats (softmax over s per destination d); recompute logits
  for (int d = tid; d < NODES; d += blockDim.x) {
    float ad = s_ad[d], mx = -3.4e38f;
    for (int s = 0; s < NODES; ++s) {
      float lg = s_as[s] + ad;
      lg = (lg > 0.0f ? lg : NEG_SLOPE * lg) * ea[s * NODES + d];
      mx = fmaxf(mx, lg);
    }
    float dn = 0.0f;
    for (int s = 0; s < NODES; ++s) {
      float lg = s_as[s] + ad;
      lg = (lg > 0.0f ? lg : NEG_SLOPE * lg) * ea[s * NODES + d];
      dn += __expf(lg - mx);
    }
    s_mx[d] = mx;
    s_dn[d] = dn + 1e-16f;
  }
  __syncthreads();

  // alpha^T into LDS (f16) + head-mean alpha to next edge_attr
  for (int i = tid; i < NODES * KP; i += blockDim.x) {
    int d = i / KP, s = i - d * KP;
    float al = 0.0f;
    if (s < NODES) {
      float lg = s_as[s] + s_ad[d];
      lg = (lg > 0.0f ? lg : NEG_SLOPE * lg) * ea[s * NODES + d];
      al = __expf(lg - s_mx[d]) / s_dn[d];
      if (MH) atomicAdd(&alpha_out[(size_t)b * E + s * NODES + d], al * inv_h);
      else alpha_out[(size_t)b * E + s * NODES + d] = al;
    }
    sAT[d * KPS + s] = (_Float16)al;
  }
  __syncthreads();

  // out[d, c] = sum_s alpha^T[d,s] * h[s,c] : M=NODES, N=32 (pad of cout), K=KP
  constexpr int MT = NODES / 16, KT = KP / 32, NT2 = 2;
  for (int t = wv; t < MT * NT2; t += nwv) {   // wave-uniform -> EXEC all-ones at WMMA
    int m0 = (t >> 1) << 4, n0 = (t & 1) << 4;
    v8f acc = {};
#pragma unroll
    for (int k = 0; k < KT; ++k) {
      v16h af = lds_load_a16(sAT, KPS, m0, k * 32, lane);
      v16h bf = lds_load_bT16(shT, KPS, k * 32, n0, lane);
      acc = __builtin_amdgcn_wmma_f32_16x16x32_f16(false, af, false, bf,
                                                   (short)0, acc, false, false);
    }
    int n = n0 + (lane & 15);
    if (n < cout) {
      int mb = m0 + ((lane >> 4) << 3);
      float* op = &out_acc[((size_t)b * NODES + mb) * cout + n];
#pragma unroll
      for (int r = 0; r < 8; ++r) {
        if (MH) atomicAdd(&op[r * cout], acc[r] * inv_h);
        else op[r * cout] = acc[r];
      }
    }
  }
}

// per-graph instance norm (affine); one block per graph, one thread per channel
__global__ void k_inorm(const float* __restrict__ x, const float* __restrict__ pre_bias,
                        const float* __restrict__ wp, const float* __restrict__ bp,
                        float* __restrict__ y, int ldy, int nn, int C) {
  int b = blockIdx.x, c = threadIdx.x;
  if (c >= C) return;
  const float* xb = x + (size_t)b * nn * C;
  float pb = pre_bias ? pre_bias[c] : 0.0f;
  float s = 0.0f, s2 = 0.0f;
  for (int n = 0; n < nn; ++n) {
    float v = xb[n * C + c] + pb;
    s += v; s2 += v * v;
  }
  float mean = s / nn;
  float var = s2 / nn - mean * mean;
  float rstd = rsqrtf(var + 1e-5f);
  float* yb = y + (size_t)b * nn * ldy;
  for (int n = 0; n < nn; ++n) {
    float v = xb[n * C + c] + pb;
    yb[n * ldy + c] = (v - mean) * rstd * wp[c] + bp[c];
  }
}

// column mean / rstd over R rows (inorm_flat stats); one block per column
__global__ void k_colstats(const float* __restrict__ x, int ld, int R,
                           float* __restrict__ mean, float* __restrict__ rstd) {
  int c = blockIdx.x, tid = threadIdx.x;
  __shared__ float ssum[256], ssq[256];
  float s = 0.0f, s2 = 0.0f;
  for (int r = tid; r < R; r += blockDim.x) {
    float v = x[(size_t)r * ld + c];
    s += v; s2 += v * v;
  }
  ssum[tid] = s; ssq[tid] = s2;
  __syncthreads();
  for (int o = blockDim.x >> 1; o > 0; o >>= 1) {
    if (tid < o) { ssum[tid] += ssum[tid + o]; ssq[tid] += ssq[tid + o]; }
    __syncthreads();
  }
  if (tid == 0) {
    float m = ssum[0] / R, v = ssq[0] / R - m * m;
    mean[c] = m;
    rstd[c] = rsqrtf(v + 1e-5f);
  }
}

__global__ void k_norm_relu(const float* __restrict__ x, int ld,
                            const float* __restrict__ mean, const float* __restrict__ rstd,
                            const float* __restrict__ wp, const float* __restrict__ bp,
                            float* __restrict__ y, int ldy, int R, int C) {
  int idx = blockIdx.x * blockDim.x + threadIdx.x;
  if (idx >= R * C) return;
  int r = idx / C, c = idx - r * C;
  float v = (x[(size_t)r * ld + c] - mean[c]) * rstd[c] * wp[c] + bp[c];
  y[(size_t)r * ldy + c] = fmaxf(v, 0.0f);
}

// dense_diff_pool per graph: softmax(s), px = s^T x, padj = s^T A s, losses
__global__ void k_diffpool(const float* __restrict__ assign, const float* __restrict__ xg,
                           const float* __restrict__ adj,
                           float* __restrict__ px, float* __restrict__ padj,
                           float* __restrict__ loss_acc) {
  int b = blockIdx.x, tid = threadIdx.x;
  __shared__ float ss[128 * 16];
  __shared__ float sx[128 * 30];
  __shared__ float st[128 * 16];
  __shared__ float red[256];
  const float* ab = assign + (size_t)b * 128 * 16;
  const float* xb = xg + (size_t)b * 128 * 30;
  const float* adjb = adj + (size_t)b * 128 * 128;

  for (int n = tid; n < 128; n += blockDim.x) {
    float m = -3.4e38f;
    for (int k = 0; k < 16; ++k) m = fmaxf(m, ab[n * 16 + k]);
    float den = 0.0f, e[16];
    for (int k = 0; k < 16; ++k) { e[k] = __expf(ab[n * 16 + k] - m); den += e[k]; }
    for (int k = 0; k < 16; ++k) ss[n * 16 + k] = e[k] / den;
  }
  for (int i = tid; i < 128 * 30; i += blockDim.x) sx[i] = xb[i];
  __syncthreads();

  for (int i = tid; i < 16 * 30; i += blockDim.x) {     // px = s^T x
    int k = i / 30, hc = i - k * 30;
    float acc = 0.0f;
    for (int n = 0; n < 128; ++n) acc += ss[n * 16 + k] * sx[n * 30 + hc];
    px[((size_t)b * 16 + k) * 30 + hc] = acc;
  }
  for (int i = tid; i < 128 * 16; i += blockDim.x) {    // t1 = A s
    int n = i / 16, l = i - n * 16;
    float acc = 0.0f;
    for (int m = 0; m < 128; ++m) acc += adjb[n * 128 + m] * ss[m * 16 + l];
    st[i] = acc;
  }
  __syncthreads();
  for (int i = tid; i < 256; i += blockDim.x) {         // padj = s^T t1
    int k = i / 16, l = i - k * 16;
    float acc = 0.0f;
    for (int n = 0; n < 128; ++n) acc += ss[n * 16 + k] * st[n * 16 + l];
    padj[(size_t)b * 256 + i] = acc;
  }

  float lk = 0.0f;                                       // ||A - s s^T||^2 part
  for (int p = tid; p < 128 * 128; p += blockDim.x) {
    int n = p >> 7, m = p & 127;
    float dot = 0.0f;
    for (int k = 0; k < 16; ++k) dot += ss[n * 16 + k] * ss[m * 16 + k];
    float df = adjb[p] - dot;
    lk += df * df;
  }
  float en = 0.0f;                                       // entropy part
  for (int i = tid; i < 128 * 16; i += blockDim.x) {
    float v = ss[i];
    en += -v * __logf(v + 1e-15f);
  }
  red[tid] = lk; __syncthreads();
  for (int o = blockDim.x >> 1; o > 0; o >>= 1) {
    if (tid < o) red[tid] += red[tid + o];
    __syncthreads();
  }
  if (tid == 0) atomicAdd(&loss_acc[0], red[0]);
  __syncthreads();
  red[tid] = en; __syncthreads();
  for (int o = blockDim.x >> 1; o > 0; o >>= 1) {
    if (tid < o) red[tid] += red[tid + o];
    __syncthreads();
  }
  if (tid == 0) atomicAdd(&loss_acc[1], red[0]);
}

// global max pool over nodes of one block output -> g[:, coloff..coloff+C)
__global__ void k_maxpool(const float* __restrict__ src, int nodes, int C,
                          float* __restrict__ g, int ldg, int coloff) {
  int idx = blockIdx.x * blockDim.x + threadIdx.x;
  if (idx >= 64 * C) return;
  int b = idx / C, c = idx - b * C;
  float m = -3.4e38f;
  for (int n = 0; n < nodes; ++n)
    m = fmaxf(m, src[((size_t)b * nodes + n) * C + c]);
  g[b * ldg + coloff + c] = m;
}

// final head: fc1 -> inorm_flat -> relu -> fc2 ; writes fc_out[128] + reg[1]
__global__ void k_head(const float* __restrict__ g,
                       const float* __restrict__ W1, const float* __restrict__ b1,
                       const float* __restrict__ wn, const float* __restrict__ bn,
                       const float* __restrict__ W2, const float* __restrict__ b2,
                       const float* __restrict__ loss_acc, float* __restrict__ out) {
  __shared__ float t[64 * 50];
  __shared__ float mean[50], rstd[50];
  int tid = threadIdx.x;
  for (int i = tid; i < 64 * 50; i += blockDim.x) {
    int r = i / 50, j = i - r * 50;
    float a = b1[j];
    for (int q = 0; q < 180; ++q) a += g[r * 180 + q] * W1[q * 50 + j];
    t[i] = a;
  }
  __syncthreads();
  for (int j = tid; j < 50; j += blockDim.x) {
    float s = 0.0f, s2 = 0.0f;
    for (int r = 0; r < 64; ++r) { float v = t[r * 50 + j]; s += v; s2 += v * v; }
    float m = s / 64.0f, v = s2 / 64.0f - m * m;
    mean[j] = m; rstd[j] = rsqrtf(v + 1e-5f);
  }
  __syncthreads();
  for (int i = tid; i < 64 * 50; i += blockDim.x) {
    int j = i % 50;
    t[i] = fmaxf((t[i] - mean[j]) * rstd[j] * wn[j] + bn[j], 0.0f);
  }
  __syncthreads();
  for (int i = tid; i < 128; i += blockDim.x) {
    int r = i >> 1, j = i & 1;
    float a = b2[j];
    for (int q = 0; q < 50; ++q) a += t[r * 50 + q] * W2[q * 2 + j];
    out[i] = a;
  }
  if (tid == 0)
    out[128] = sqrtf(loss_acc[0]) / (64.0f * 128.0f * 128.0f) + loss_acc[1] / (64.0f * 128.0f);
}

// ---------------------------------------------------------------------------
extern "C" void kernel_launch(void* const* d_in, const int* in_sizes, int n_in,
                              void* d_out, int out_size, void* d_ws, size_t ws_size,
                              hipStream_t stream) {
  (void)in_sizes; (void)n_in; (void)out_size; (void)ws_size;
  const int B = 64, N = 128, KP = 16, H = 30, HEADS = 5;
  const int NT = B * N;                  // 8192
  const size_t Ebig = (size_t)B * N * N; // 1,048,576

  const float* x = (const float*)d_in[0];
  // d_in[1] = edge_index: unused (topology is fixed fully-connected, src-major)
  const float* edge_attr = (const float*)d_in[2];

  // Params flattened in jax tree order (dict keys sorted alphabetically):
  // conv1, conv2, fc1, fc2, fc_norm, norm1, norm2, pool_conv, pool_fc1,
  // pool_fc2, pool_fc_norm, pool_norm. EGAT dicts: W, att_dst, att_src, bias.
  // _lin: W, b. _norm: b, w.
  struct Egat { const float *W, *adst, *asrc, *bias; };
  struct Norm { const float *b, *w; };
  int pi = 3;
  auto f = [&](void) { return (const float*)d_in[pi++]; };
  Egat conv1[3], conv2[3], poolc[3];
  Norm norm1[3], norm2[3], pooln[3];
  for (int i = 0; i < 3; ++i) { conv1[i].W = f(); conv1[i].adst = f(); conv1[i].asrc = f(); conv1[i].bias = f(); }
  for (int i = 0; i < 3; ++i) { conv2[i].W = f(); conv2[i].adst = f(); conv2[i].asrc = f(); conv2[i].bias = f(); }
  const float *fc1W = f(), *fc1b = f();
  const float *fc2W = f(), *fc2b = f();
  const float *fcn_b = f(), *fcn_w = f();
  for (int i = 0; i < 3; ++i) { norm1[i].b = f(); norm1[i].w = f(); }
  for (int i = 0; i < 3; ++i) { norm2[i].b = f(); norm2[i].w = f(); }
  for (int i = 0; i < 3; ++i) { poolc[i].W = f(); poolc[i].adst = f(); poolc[i].asrc = f(); poolc[i].bias = f(); }
  const float *pfc1W = f(), *pfc1b = f();
  const float *pfc2W = f(), *pfc2b = f();
  const float *pfcn_b = f(), *pfcn_w = f();
  for (int i = 0; i < 3; ++i) { pooln[i].b = f(); pooln[i].w = f(); }

  // ---- workspace bump allocator (~23 MB of f32) ----
  float* wsp = (float*)d_ws;
  auto alloc = [&](size_t n) { float* p = wsp; wsp += n; return p; };
  float* hbuf = alloc((size_t)NT * HEADS * H);
  float* asb = alloc((size_t)NT * HEADS);
  float* adb = alloc((size_t)NT * HEADS);
  float* aA = alloc(Ebig);
  float* aB = alloc(Ebig);
  float* agg = alloc((size_t)NT * H);
  float* o1[3] = { alloc((size_t)NT * H), alloc((size_t)NT * H), alloc((size_t)NT * H) };
  float* pc = alloc((size_t)NT * 76);
  float* tbuf = alloc((size_t)NT * 50);
  float* assign = alloc((size_t)NT * KP);
  float* h2 = alloc((size_t)B * KP * H);
  float* padjb = alloc((size_t)B * KP * KP);
  float* o2[3] = { alloc((size_t)B * KP * H), alloc((size_t)B * KP * H), alloc((size_t)B * KP * H) };
  float* meanb = alloc(64);
  float* rstdb = alloc(64);
  float* gbuf = alloc((size_t)B * 180);
  float* lossb = alloc(2);

  // ---- one EGAT layer + instance norm ----
  auto run_egat = [&](const float* xin, int ldx, int cin, int heads, int cout,
                      int nodes, int ngraphs, const Egat& e, const Norm& nm,
                      const float* ea_in, float* ea_out, float* out_norm, int ldo) {
    int nt = nodes * ngraphs;
    size_t Eg = (size_t)ngraphs * nodes * nodes;
    k_linear<<<cdiv((size_t)nt * heads * cout, 256), 256, 0, stream>>>(
        xin, ldx, e.W, nullptr, hbuf, heads * cout, nt, cin, heads * cout);
    k_attn_dots<<<cdiv((size_t)nt * heads, 256), 256, 0, stream>>>(
        hbuf, e.asrc, e.adst, asb, adb, nt, heads, cout);
    if (heads > 1) {  // atomically-accumulated head means need zero init
      k_fill<<<cdiv(Eg, 256), 256, 0, stream>>>(ea_out, 0.0f, Eg);
      k_fill<<<cdiv((size_t)nt * cout, 256), 256, 0, stream>>>(agg, 0.0f, (size_t)nt * cout);
      k_egat<128, true><<<dim3(ngraphs, heads), 256, 0, stream>>>(hbuf, asb, adb, ea_in, ea_out, agg, heads, cout);
    } else if (nodes == 128) {
      k_egat<128, false><<<dim3(ngraphs, 1), 256, 0, stream>>>(hbuf, asb, adb, ea_in, ea_out, agg, 1, cout);
    } else {
      k_egat<16, false><<<dim3(ngraphs, 1), 128, 0, stream>>>(hbuf, asb, adb, ea_in, ea_out, agg, 1, cout);
    }
    k_inorm<<<ngraphs, 32, 0, stream>>>(agg, e.bias, nm.w, nm.b, out_norm, ldo, nodes, cout);
  };

  // conv1 path (out_all[0..2])
  run_egat(x, 11, 11, HEADS, H, N, B, conv1[0], norm1[0], edge_attr, aA, o1[0], H);
  run_egat(o1[0], H, H, 1, H, N, B, conv1[1], norm1[1], aA, aB, o1[1], H);
  run_egat(o1[1], H, H, 1, H, N, B, conv1[2], norm1[2], aB, aA, o1[2], H);

  // pool path -> pc = concat(p0,p1,p2) [NT, 76] (written in-place via ld=76)
  run_egat(x, 11, 11, 1, H, N, B, poolc[0], pooln[0], edge_attr, aB, pc + 0, 76);
  run_egat(pc + 0, 76, H, 1, H, N, B, poolc[1], pooln[1], aB, aA, pc + 30, 76);
  run_egat(pc + 30, 76, H, 1, KP, N, B, poolc[2], pooln[2], aA, aB, pc + 60, 76);

  // pool MLP: pc @ W1 + b1 -> inorm_flat -> relu -> @ W2 + b2 = assignment
  k_linear<<<cdiv((size_t)NT * 50, 256), 256, 0, stream>>>(pc, 76, pfc1W, pfc1b, tbuf, 50, NT, 76, 50);
  k_colstats<<<50, 256, 0, stream>>>(tbuf, 50, NT, meanb, rstdb);
  k_norm_relu<<<cdiv((size_t)NT * 50, 256), 256, 0, stream>>>(tbuf, 50, meanb, rstdb, pfcn_w, pfcn_b, tbuf, 50, NT, 50);
  k_linear<<<cdiv((size_t)NT * KP, 256), 256, 0, stream>>>(tbuf, 50, pfc2W, pfc2b, assign, KP, NT, 50, KP);

  // dense diff pool (softmax, px, padj, link+entropy losses)
  k_fill<<<1, 32, 0, stream>>>(lossb, 0.0f, 2);
  k_diffpool<<<B, 256, 0, stream>>>(assign, o1[2], edge_attr, h2, padjb, lossb);

  // conv2 path on pooled graphs (out_all[3..5])
  run_egat(h2, H, H, 1, H, KP, B, conv2[0], norm2[0], padjb, aA, o2[0], H);
  run_egat(o2[0], H, H, 1, H, KP, B, conv2[1], norm2[1], aA, aB, o2[1], H);
  run_egat(o2[1], H, H, 1, H, KP, B, conv2[2], norm2[2], aB, aA, o2[2], H);

  // global max pool -> g [B, 180]
  k_maxpool<<<cdiv((size_t)B * H, 256), 256, 0, stream>>>(o1[0], N, H, gbuf, 180, 0);
  k_maxpool<<<cdiv((size_t)B * H, 256), 256, 0, stream>>>(o1[1], N, H, gbuf, 180, 30);
  k_maxpool<<<cdiv((size_t)B * H, 256), 256, 0, stream>>>(o1[2], N, H, gbuf, 180, 60);
  k_maxpool<<<cdiv((size_t)B * H, 256), 256, 0, stream>>>(o2[0], KP, H, gbuf, 180, 90);
  k_maxpool<<<cdiv((size_t)B * H, 256), 256, 0, stream>>>(o2[1], KP, H, gbuf, 180, 120);
  k_maxpool<<<cdiv((size_t)B * H, 256), 256, 0, stream>>>(o2[2], KP, H, gbuf, 180, 150);

  // head MLP + reg
  k_head<<<1, 256, 0, stream>>>(gbuf, fc1W, fc1b, fcn_w, fcn_b, fc2W, fc2b, lossb, (float*)d_out);
}
